// LocalWindowAttention_10067403342471
// MI455X (gfx1250) — matrix operand order repaired and checked by reading, not compiled, not run on hardware
//
#include <hip/hip_runtime.h>
#include <hip/hip_bf16.h>

typedef _Float16 v16h __attribute__((ext_vector_type(16)));
typedef _Float16 v8h  __attribute__((ext_vector_type(8)));
typedef float    v8f  __attribute__((ext_vector_type(8)));

#define CCH   192     // channels
#define HH    224
#define WWD   224
#define WSZ   7
#define NTOK  49      // tokens per window
#define MPAD  64      // padded token rows (4 x 16)
#define LDX   200     // leading dim (halfs) for 64x192 buffers; 400B rows, 16B aligned
#define LDV   72      // leading dim for 64-col buffers (attn, V^T); 144B rows, 16B aligned
#define LDSC  65      // f32 score row stride
#define LDOF  193     // f32 output-staging row stride (odd DW -> conflict-free t-strided reads)

// ---- WMMA fragment loaders (wave32, ISA 7.12.2 layouts) --------------------
// A: 16x32 f16 tile, row-major source, leading dim ld (halfs).
__device__ __forceinline__ v16h ldA(const _Float16* base, int ld, int m0, int k0) {
  const int lane = threadIdx.x & 31;
  const _Float16* p = base + (m0 + (lane & 15)) * ld + k0 + ((lane >> 4) << 3);
  v8h lo = *(const v8h*)(p);
  v8h hi = *(const v8h*)(p + 16);
  v16h r;
#pragma unroll
  for (int i = 0; i < 8; ++i) { r[i] = lo[i]; r[i + 8] = hi[i]; }
  return r;
}

// B: 32x16 f16 tile taken from an N-major [N][K] buffer (i.e. B^T row-major).
__device__ __forceinline__ v16h ldB(const _Float16* base, int ld, int n0, int k0) {
  const int lane = threadIdx.x & 31;
  const _Float16* p = base + (n0 + (lane & 15)) * ld + k0 + ((lane >> 4) << 3);
  v8h lo = *(const v8h*)(p);
  v8h hi = *(const v8h*)(p + 16);
  v16h r;
#pragma unroll
  for (int i = 0; i < 8; ++i) { r[i] = lo[i]; r[i + 8] = hi[i]; }
  return r;
}

// ---- weight f32 -> f16 conversion (runs once; result lives in L2) ----------
__global__ void lwa_cvt_f16(const float* __restrict__ s, _Float16* __restrict__ d, int n) {
  int i = blockIdx.x * blockDim.x + threadIdx.x;
  if (i < n) d[i] = (_Float16)s[i];
}

// ---- fused per-window attention --------------------------------------------
__global__ __launch_bounds__(256) void lwa_fused(
    const float* __restrict__ x,
    const _Float16* __restrict__ Wq, const _Float16* __restrict__ Wk,
    const _Float16* __restrict__ Wv, const _Float16* __restrict__ Wo,
    const float* __restrict__ bq, const float* __restrict__ bk,
    const float* __restrict__ bv, const float* __restrict__ bo,
    float* __restrict__ out)
{
  extern __shared__ char smem[];
  _Float16* Xh = (_Float16*)smem;           // 64 x LDX  (x window; later reused as attn@V out)
  _Float16* Qh = Xh + MPAD * LDX;           // 64 x LDX  row-major (A operand)
  _Float16* Kh = Qh + MPAD * LDX;           // 64 x LDX  token-major == N x K for scores
  _Float16* Vt = Kh + MPAD * LDX;           // 192 x LDV V^T (channel-major == N x K)
  float*    Sf = (float*)(Vt + CCH * LDV);  // 64 x LDSC f32 scores
  _Float16* Ah = (_Float16*)(Sf + MPAD * LDSC); // 64 x LDV attn probs (A operand)
  float*    Of = (float*)Qh;                // 64 x LDOF f32 out staging (Qh/Kh dead by then)

  const int tid  = threadIdx.x;
  const int wave = tid >> 5;
  const int lane = tid & 31;

  const int blk = blockIdx.x;               // 8192 windows
  const int b   = blk >> 10;
  const int wh  = (blk >> 5) & 31;
  const int ww  = blk & 31;
  const int h0  = wh * WSZ, w0 = ww * WSZ;

  // Phase 1: stage x window into LDS as f16. t is the inner index so adjacent
  // lanes read contiguous w (28B runs) -> coalesced HBM traffic.
  for (int i = tid; i < MPAD * CCH; i += 256) {
    int c = i >> 6;          // 0..191
    int t = i & 63;          // 0..63
    float v = 0.f;
    if (t < NTOK) {
      int hh = h0 + t / WSZ, wl = w0 + t % WSZ;
      v = x[((b * CCH + c) * HH + hh) * WWD + wl];
    }
    Xh[t * LDX + c] = (_Float16)v;
  }
  __syncthreads();

  // Phase 2: Q/K/V projections. 3 matrices x (4 M-tiles x 12 N-tiles) = 144 tiles.
  for (int it = 0; it < 18; ++it) {
    int idx = wave + (it << 3);
    int mat = idx / 48;
    int rem = idx - mat * 48;
    int mt = rem & 3;
    int nt = rem >> 2;
    const _Float16* Wp = (mat == 0) ? Wq : (mat == 1) ? Wk : Wv;
    const float*    bp = (mat == 0) ? bq : (mat == 1) ? bk : bv;
    v8f acc = {};
#pragma unroll
    for (int k = 0; k < 6; ++k) {
      v16h a  = ldA(Xh, LDX, mt * 16, k * 32);
      v16h bb = ldB(Wp, CCH, nt * 16, k * 32);   // W row-major [o][c] == N x K
      acc = __builtin_amdgcn_wmma_f32_16x16x32_f16(false, a, false, bb,
                                                   (short)0, acc, false, false);
    }
    int n  = nt * 16 + (lane & 15);
    int mb = mt * 16 + ((lane >> 4) << 3);
    float bias = bp[n];
    if (mat == 2) {            // V stored transposed: Vt[c][t]
#pragma unroll
      for (int r = 0; r < 8; ++r) Vt[n * LDV + mb + r] = (_Float16)(acc[r] + bias);
    } else {
      _Float16* dst = (mat == 0) ? Qh : Kh;
#pragma unroll
      for (int r = 0; r < 8; ++r) dst[(mb + r) * LDX + n] = (_Float16)(acc[r] + bias);
    }
  }
  __syncthreads();

  // Phase 3: scores = Q K^T / sqrt(C). 4x4 tiles, 2 per wave.
  const float scale = 0.07216878364870322f;   // 1/sqrt(192)
  for (int it = 0; it < 2; ++it) {
    int idx = wave + (it << 3);
    int mt = idx >> 2, ut = idx & 3;
    v8f acc = {};
#pragma unroll
    for (int k = 0; k < 6; ++k) {
      v16h a  = ldA(Qh, LDX, mt * 16, k * 32);
      v16h bb = ldB(Kh, LDX, ut * 16, k * 32);  // Kh token-major == N x K
      acc = __builtin_amdgcn_wmma_f32_16x16x32_f16(false, a, false, bb,
                                                   (short)0, acc, false, false);
    }
    int n  = ut * 16 + (lane & 15);
    int mb = mt * 16 + ((lane >> 4) << 3);
#pragma unroll
    for (int r = 0; r < 8; ++r) Sf[(mb + r) * LDSC + n] = acc[r] * scale;
  }
  __syncthreads();

  // Phase 4: row softmax over the 49 valid keys, 4 threads per row with
  // cross-lane (__shfl_xor) reductions; padded cols -> 0.
  {
    int t = tid >> 2;        // 0..63
    int q = tid & 3;
    float m = -1e30f;
    for (int u = q; u < NTOK; u += 4) m = fmaxf(m, Sf[t * LDSC + u]);
    m = fmaxf(m, __shfl_xor(m, 1, 32));
    m = fmaxf(m, __shfl_xor(m, 2, 32));
    float s = 0.f;
    for (int u = q; u < NTOK; u += 4) s += __expf(Sf[t * LDSC + u] - m);
    s += __shfl_xor(s, 1, 32);
    s += __shfl_xor(s, 2, 32);
    float inv = 1.f / s;
    for (int u = q; u < NTOK; u += 4)
      Ah[t * LDV + u] = (_Float16)(__expf(Sf[t * LDSC + u] - m) * inv);
    for (int u = NTOK + q; u < MPAD; u += 4) Ah[t * LDV + u] = (_Float16)0.f;
  }
  __syncthreads();

  // Phase 5: OW = attn x V  (4x12 tiles, K=64). Reuse Xh as destination.
  for (int it = 0; it < 6; ++it) {
    int idx = wave + (it << 3);
    int mt = idx & 3, nt = idx >> 2;
    v8f acc = {};
#pragma unroll
    for (int k = 0; k < 2; ++k) {
      v16h a  = ldA(Ah, LDV, mt * 16, k * 32);
      v16h bb = ldB(Vt, LDV, nt * 16, k * 32);
      acc = __builtin_amdgcn_wmma_f32_16x16x32_f16(false, a, false, bb,
                                                   (short)0, acc, false, false);
    }
    int n  = nt * 16 + (lane & 15);
    int mb = mt * 16 + ((lane >> 4) << 3);
#pragma unroll
    for (int r = 0; r < 8; ++r) Xh[(mb + r) * LDX + n] = (_Float16)acc[r];
  }
  __syncthreads();

  // Phase 6: out = OW Wo^T + bo, staged to LDS f32 (coalesced writeout after).
  for (int it = 0; it < 6; ++it) {
    int idx = wave + (it << 3);
    int mt = idx & 3, nt = idx >> 2;
    v8f acc = {};
#pragma unroll
    for (int k = 0; k < 6; ++k) {
      v16h a  = ldA(Xh, LDX, mt * 16, k * 32);
      v16h bb = ldB(Wo, CCH, nt * 16, k * 32);
      acc = __builtin_amdgcn_wmma_f32_16x16x32_f16(false, a, false, bb,
                                                   (short)0, acc, false, false);
    }
    int n  = nt * 16 + (lane & 15);
    int mb = mt * 16 + ((lane >> 4) << 3);
    float bias = bo[n];
#pragma unroll
    for (int r = 0; r < 8; ++r) Of[(mb + r) * LDOF + n] = acc[r] + bias;
  }
  __syncthreads();

  // Phase 7: coalesced writeout; adjacent lanes walk contiguous w.
  for (int i = tid; i < NTOK * CCH; i += 256) {
    int c = i / NTOK;
    int t = i - c * NTOK;
    int hh = h0 + t / WSZ, wl = w0 + t % WSZ;
    out[((b * CCH + c) * HH + hh) * WWD + wl] = Of[t * LDOF + c];
  }
}

extern "C" void kernel_launch(void* const* d_in, const int* in_sizes, int n_in,
                              void* d_out, int out_size, void* d_ws, size_t ws_size,
                              hipStream_t stream) {
  const float* x  = (const float*)d_in[0];
  const float* Wq = (const float*)d_in[1];
  const float* bq = (const float*)d_in[2];
  const float* Wk = (const float*)d_in[3];
  const float* bk = (const float*)d_in[4];
  const float* Wv = (const float*)d_in[5];
  const float* bv = (const float*)d_in[6];
  const float* Wo = (const float*)d_in[7];
  const float* bo = (const float*)d_in[8];
  float* out = (float*)d_out;

  _Float16* wh = (_Float16*)d_ws;           // 4 x 36864 halfs = 288 KB
  const int NW = CCH * CCH;
  lwa_cvt_f16<<<(NW + 255) / 256, 256, 0, stream>>>(Wq, wh,          NW);
  lwa_cvt_f16<<<(NW + 255) / 256, 256, 0, stream>>>(Wk, wh + NW,     NW);
  lwa_cvt_f16<<<(NW + 255) / 256, 256, 0, stream>>>(Wv, wh + 2 * NW, NW);
  lwa_cvt_f16<<<(NW + 255) / 256, 256, 0, stream>>>(Wo, wh + 3 * NW, NW);

  const size_t smem = (size_t)(3 * MPAD * LDX * 2 + CCH * LDV * 2 +
                               MPAD * LDSC * 4 + MPAD * LDV * 2);  // 130304 B
  (void)hipFuncSetAttribute((const void*)lwa_fused,
                            hipFuncAttributeMaxDynamicSharedMemorySize, (int)smem);
  lwa_fused<<<dim3(8192), dim3(256), smem, stream>>>(
      x, wh, wh + NW, wh + 2 * NW, wh + 3 * NW, bq, bk, bv, bo, out);
}
